// VAE_84430467105217
// MI455X (gfx1250) — matrix operand order repaired and checked by reading, not compiled, not run on hardware
//
#include <hip/hip_runtime.h>

// ---------------------------------------------------------------------------
// MI455X (gfx1250, wave32) mixture-VAE forward.
//  * Big GEMMs: V_WMMA_F32_16X16X32_BF16 (bf16 in, f32 acc). ~100 GFLOP vs
//    ~0.5 GB traffic -> bf16 WMMA sits at the 23.3 TB/s HBM roofline (~21 us);
//    f32 WMMA (16x16x4) would be 8x slower on math and compute-bound.
//  * Tiles move via GLOBAL_LOAD_ASYNC_TO_LDS_B128 (ASYNCcnt) with LDS
//    double-buffering: next tile's async copies are in flight while the
//    current tile runs through the WMMAs.
//  * K dims padded to mult-of-32 (784->800, 400->416) with zeroed weight pads
//    so async copies are unconditional (exact asynccnt discipline, no tails).
//  * var/inv(Q) elided: z = Q * (eigen .* solve(Q, eps)) + mu.
// ---------------------------------------------------------------------------

typedef __attribute__((ext_vector_type(16))) __bf16 v16bf;
typedef __attribute__((ext_vector_type(8)))  __bf16 v8bf;
typedef __attribute__((ext_vector_type(4)))  __bf16 v4bf;
typedef __attribute__((ext_vector_type(8)))  float  v8f;
typedef __attribute__((ext_vector_type(4)))  float  v4f;
typedef __attribute__((ext_vector_type(4)))  int    v4i;

#define BM 128
#define BN 128
#define BK 32
#define LDST 40   // LDS row stride (bf16): 80B rows -> b128 accesses stay 16B aligned

enum { ACT_NONE = 0, ACT_EXP = 1, ACT_SIGMOID = 2, ACT_RELU_BF16 = 3 };

// ---- CDNA5 async global->LDS copy (16B per lane), ASYNCcnt-tracked --------
// Builtin signature (from hipcc diagnostic): arg0 is AS1 (global) v4i*,
// arg1 is the LDS destination, then (offset, cpol).
static __device__ __forceinline__ void async_copy16(const __bf16* __restrict__ g, __bf16* l) {
#if __has_builtin(__builtin_amdgcn_global_load_async_to_lds_b128)
    __builtin_amdgcn_global_load_async_to_lds_b128(
        (__attribute__((address_space(1))) v4i*)g,
        (__attribute__((address_space(3))) v4i*)l, 0, 0);
#else
    unsigned loff = (unsigned)(unsigned long long)(__attribute__((address_space(3))) __bf16*)l;
    asm volatile("global_load_async_to_lds_b128 %0, %1, off"
                 :
                 : "v"(loff), "v"((const void*)g)
                 : "memory");
#endif
}

template <int N>
static __device__ __forceinline__ void wait_async() {
#if __has_builtin(__builtin_amdgcn_s_wait_asynccnt)
    __builtin_amdgcn_s_wait_asynccnt(N);
#else
    asm volatile("s_wait_asynccnt %0" : : "i"(N) : "memory");
#endif
}

// Issue one double-buffer stage: A tile [BM x BK] and B tile [BN x BK], both
// row-major with stride ld (= padded K). Exactly 4 async ops per thread
// (rows clamped, never masked) -> wait_async<4> retires the previous stage.
static __device__ __forceinline__ void issue_tile(__bf16* sA, __bf16* sB,
                                                  const __bf16* __restrict__ gA,
                                                  const __bf16* __restrict__ gB,
                                                  int m0, int Mrows, int n0, int Nrows,
                                                  int k0, int ld, int tid) {
#pragma unroll
    for (int it = 0; it < (BM * BK) / (256 * 8); ++it) {   // 2 iters
        int flat = (it * 256 + tid) * 8;
        int r = flat / BK, k = flat % BK;
        int ga = m0 + r; ga = ga < Mrows ? ga : Mrows - 1;  // clamp: OOB rows load
        int gb = n0 + r; gb = gb < Nrows ? gb : Nrows - 1;  // garbage, never stored
        async_copy16(gA + (size_t)ga * ld + k0 + k, sA + r * LDST + k);
        async_copy16(gB + (size_t)gb * ld + k0 + k, sB + r * LDST + k);
    }
}

// A fragment (16x32 bf16, ISA 7.12.2): lanes 0-15 hold K {0..7,16..23},
// lanes 16-31 hold K {8..15,24..31}; row = lane%16.
static __device__ __forceinline__ v16bf load_frag_a(const __bf16* s, int row, int laneHalf) {
    const __bf16* p = s + row * LDST + laneHalf * 8;
    v8bf lo = *(const v8bf*)p;
    v8bf hi = *(const v8bf*)(p + 16);
    return __builtin_shufflevector(lo, hi, 0, 1, 2, 3, 4, 5, 6, 7, 8, 9, 10, 11, 12, 13, 14, 15);
}

// B fragment (32x16 bf16): lane holds column n = lane%16; lanes 0-15: K {0..7,16..23},
// lanes 16-31: K {8..15,24..31} (matches A). Bt stored [n][k].
static __device__ __forceinline__ v16bf load_frag_b(const __bf16* s, int row, int laneHalf) {
    const __bf16* p = s + row * LDST + laneHalf * 8;
    v8bf lo = *(const v8bf*)p;
    v8bf hi = *(const v8bf*)(p + 16);
    return __builtin_shufflevector(lo, hi, 0, 1, 2, 3, 4, 5, 6, 7, 8, 9, 10, 11, 12, 13, 14, 15);
}

// C = act(A[M,Kp] @ W + bias).  A: bf16 [M][Kp]; Bt: bf16 [N][Kp] (W transposed,
// k >= K zero-padded).  256 threads = 8 waves (2x4), 64x32 per wave -> 8 WMMA/k-step.
// ldc: output stride (f32 outs: ldc==N; bf16 aux: padded, cols N..ldc zeroed).
template <int ACT>
__global__ __launch_bounds__(256) void gemm_bias_act(const __bf16* __restrict__ Ap,
                                                     const __bf16* __restrict__ Bt,
                                                     const float* __restrict__ bias,
                                                     void* __restrict__ Cp,
                                                     int M, int N, int Kp, int ldc) {
    __shared__ __bf16 lds[2 * (BM + BN) * LDST];   // 40 KB, double-buffered
    __bf16* A0 = lds;
    __bf16* B0 = lds + BM * LDST;
    __bf16* A1 = lds + (BM + BN) * LDST;
    __bf16* B1 = A1 + BM * LDST;

    const int tid = threadIdx.x;
    const int m0 = blockIdx.y * BM;
    const int n0 = blockIdx.x * BN;
    const int wave = tid >> 5, lane = tid & 31;
    const int wm = wave & 1, wn = wave >> 1;
    const int laneRow = lane & 15, laneHalf = lane >> 4;

    v8f acc[4][2];
#pragma unroll
    for (int fm = 0; fm < 4; ++fm)
#pragma unroll
        for (int fn = 0; fn < 2; ++fn)
#pragma unroll
            for (int i = 0; i < 8; ++i) acc[fm][fn][i] = 0.f;

    const int nk = Kp / BK;
    issue_tile(A0, B0, Ap, Bt, m0, M, n0, N, 0, Kp, tid);

    for (int kt = 0; kt < nk; ++kt) {
        __bf16* curA = (kt & 1) ? A1 : A0;
        __bf16* curB = (kt & 1) ? B1 : B0;
        if (kt + 1 < nk) {
            issue_tile((kt & 1) ? A0 : A1, (kt & 1) ? B0 : B1, Ap, Bt,
                       m0, M, n0, N, (kt + 1) * BK, Kp, tid);
            wait_async<4>();   // oldest 4 (current tile) retired; next 4 in flight
        } else {
            wait_async<0>();
        }
        __syncthreads();       // all waves' slices of current tile landed

        v16bf af[4], bfr[2];
#pragma unroll
        for (int fm = 0; fm < 4; ++fm)
            af[fm] = load_frag_a(curA, wm * 64 + fm * 16 + laneRow, laneHalf);
#pragma unroll
        for (int fn = 0; fn < 2; ++fn)
            bfr[fn] = load_frag_b(curB, wn * 32 + fn * 16 + laneRow, laneHalf);

#pragma unroll
        for (int fm = 0; fm < 4; ++fm)
#pragma unroll
            for (int fn = 0; fn < 2; ++fn)
                acc[fm][fn] = __builtin_amdgcn_wmma_f32_16x16x32_bf16(
                    false, af[fm], false, bfr[fn], (short)0, acc[fm][fn], false, false);
        __syncthreads();       // everyone done reading before this buffer is overwritten
    }

    // Epilogue: C/D layout (ISA 7.12.2): VGPR i -> M = i + 8*laneHalf, N = lane%16.
#pragma unroll
    for (int fn = 0; fn < 2; ++fn) {
        int col = n0 + wn * 32 + fn * 16 + laneRow;
        if (col >= ldc) continue;
        bool valid = col < N;
        float bb = valid ? bias[col] : 0.f;
#pragma unroll
        for (int fm = 0; fm < 4; ++fm) {
#pragma unroll
            for (int i = 0; i < 8; ++i) {
                int row = m0 + wm * 64 + fm * 16 + laneHalf * 8 + i;
                float v = acc[fm][fn][i] + bb;
                if (ACT == ACT_RELU_BF16) {
                    ((__bf16*)Cp)[(size_t)row * ldc + col] =
                        valid ? (__bf16)fmaxf(v, 0.f) : (__bf16)0.f;   // zero the K-pad
                } else if (valid) {
                    size_t o = (size_t)row * N + col;
                    if (ACT == ACT_NONE)
                        ((float*)Cp)[o] = v;
                    else if (ACT == ACT_EXP)
                        ((float*)Cp)[o] = __expf(v);
                    else
                        ((float*)Cp)[o] = 1.f / (1.f + __expf(-v));
                }
            }
        }
    }
}

// x (f32 [B,784]) -> xb (bf16 [B,800], K-pad zeroed).
__global__ void cvt_x_kernel(const float* __restrict__ x, __bf16* __restrict__ xb,
                             int rows, int K, int Kp) {
    int i4 = (blockIdx.x * 256 + threadIdx.x) * 4;
    if (i4 >= rows * Kp) return;
    int r = i4 / Kp, k = i4 % Kp;
    v4bf h;
    if (k + 4 <= K) {
        v4f f = *(const v4f*)(x + (size_t)r * K + k);
#pragma unroll
        for (int i = 0; i < 4; ++i) h[i] = (__bf16)f[i];
    } else {
#pragma unroll
        for (int i = 0; i < 4; ++i) h[i] = (__bf16)0.f;
    }
    *(v4bf*)(xb + (size_t)r * Kp + k) = h;
}

// WT[n*Kout + k] = bf16(W[k*N + n]); rows k >= K zero (pads the GEMM K dim).
__global__ void transpose_cvt_kernel(const float* __restrict__ W, __bf16* __restrict__ WT,
                                     int K, int N, int Kout) {
    int i = blockIdx.x * 256 + threadIdx.x;
    if (i >= N * Kout) return;
    int n = i / Kout, k = i % Kout;
    WT[i] = (k < K) ? (__bf16)W[(size_t)k * N + n] : (__bf16)0.f;
}

// logits = h1 @ Wa + ba (N=5: too small for WMMA); softmax; inverse-CDF pick.
__global__ void mixture_select_kernel(const __bf16* __restrict__ h1b, const float* __restrict__ Wa,
                                      const float* __restrict__ ba, const float* __restrict__ u,
                                      float* __restrict__ aOut, int* __restrict__ idxOut,
                                      int Bsz, int ldh) {
    int gw = (int)((blockIdx.x * (size_t)blockDim.x + threadIdx.x) >> 5);
    int lane = threadIdx.x & 31;
    if (gw >= Bsz) return;
    const __bf16* h = h1b + (size_t)gw * ldh;
    float a0 = 0.f, a1 = 0.f, a2 = 0.f, a3 = 0.f, a4 = 0.f;
    for (int k = lane; k < 400; k += 32) {
        float hv = (float)h[k];
        const float* wr = Wa + k * 5;
        a0 += hv * wr[0]; a1 += hv * wr[1]; a2 += hv * wr[2];
        a3 += hv * wr[3]; a4 += hv * wr[4];
    }
#pragma unroll
    for (int off = 16; off > 0; off >>= 1) {
        a0 += __shfl_down(a0, off); a1 += __shfl_down(a1, off);
        a2 += __shfl_down(a2, off); a3 += __shfl_down(a3, off);
        a4 += __shfl_down(a4, off);
    }
    if (lane == 0) {
        float l[5] = {a0 + ba[0], a1 + ba[1], a2 + ba[2], a3 + ba[3], a4 + ba[4]};
        float mx = fmaxf(fmaxf(fmaxf(l[0], l[1]), fmaxf(l[2], l[3])), l[4]);
        float e[5], s = 0.f;
#pragma unroll
        for (int n = 0; n < 5; ++n) { e[n] = __expf(l[n] - mx); s += e[n]; }
        float inv = 1.f / s, uu = u[gw], cdf = 0.f;
        int id = 0; bool found = false;
#pragma unroll
        for (int n = 0; n < 5; ++n) {
            float an = e[n] * inv;
            aOut[(size_t)gw * 5 + n] = an;
            cdf += an;
            if (!found && uu < cdf) { id = n; found = true; }
        }
        idxOut[gw] = id;   // argmax(all False) == 0, matches jnp.argmax
    }
}

// Per row: gather selected (Q, mu, eigen), partial-pivot solve Q y = eps,
// z = Q (eigen .* y) + mu, stored bf16 zero-padded to K=32.
__global__ void solve_z_kernel(const float* __restrict__ qO, const float* __restrict__ muO,
                               const float* __restrict__ eO, const int* __restrict__ idx,
                               const float* __restrict__ eps, __bf16* __restrict__ zb) {
    __shared__ float aug[20][24];
    __shared__ float Qm[20][20];
    __shared__ float ev[20], yv[20], muv[20];
    __shared__ int piv;
    const int b = blockIdx.x, t = threadIdx.x;
    const int id = idx[b];
    const float* q = qO + (size_t)b * 2000 + (size_t)id * 400;
    for (int i = t; i < 400; i += 32) {
        float v = q[i];
        Qm[i / 20][i % 20] = v;
        aug[i / 20][i % 20] = v;
    }
    if (t < 20) {
        ev[t] = eO[(size_t)b * 100 + id * 20 + t];
        muv[t] = muO[(size_t)b * 100 + id * 20 + t];
        aug[t][20] = eps[(size_t)b * 20 + t];
    }
    __syncthreads();
    for (int k = 0; k < 20; ++k) {
        if (t == 0) {
            int p = k; float best = fabsf(aug[k][k]);
            for (int r = k + 1; r < 20; ++r) {
                float v = fabsf(aug[r][k]);
                if (v > best) { best = v; p = r; }
            }
            piv = p;
        }
        __syncthreads();
        int p = piv;
        if (p != k && t <= 20) {
            float tmp = aug[k][t]; aug[k][t] = aug[p][t]; aug[p][t] = tmp;
        }
        __syncthreads();
        if (t > k && t < 20) {
            float f = aug[t][k] / aug[k][k];
            for (int c = k; c <= 20; ++c) aug[t][c] -= f * aug[k][c];
        }
        __syncthreads();
    }
    if (t == 0) {
        for (int k = 19; k >= 0; --k) {
            float s = aug[k][20];
            for (int c = k + 1; c < 20; ++c) s -= aug[k][c] * yv[c];
            yv[k] = s / aug[k][k];
        }
    }
    __syncthreads();
    if (t < 20) {
        float s = muv[t];
        for (int j = 0; j < 20; ++j) s += Qm[t][j] * ev[j] * yv[j];
        zb[(size_t)b * 32 + t] = (__bf16)s;
    } else {
        zb[(size_t)b * 32 + t] = (__bf16)0.f;
    }
}

extern "C" void kernel_launch(void* const* d_in, const int* in_sizes, int n_in,
                              void* d_out, int out_size, void* d_ws, size_t ws_size,
                              hipStream_t stream) {
    (void)in_sizes; (void)n_in; (void)out_size; (void)ws_size;
    const int B = 32768;
    const int K1P = 800, K2P = 416;          // padded K dims (mult of 32)
    const float* x   = (const float*)d_in[0];
    const float* u   = (const float*)d_in[1];
    const float* eps = (const float*)d_in[2];
    const float* W1  = (const float*)d_in[3];  const float* b1  = (const float*)d_in[4];
    const float* Wmu = (const float*)d_in[5];  const float* bmu = (const float*)d_in[6];
    const float* WQ  = (const float*)d_in[7];  const float* bQ  = (const float*)d_in[8];
    const float* Wa  = (const float*)d_in[9];  const float* ba  = (const float*)d_in[10];
    const float* We  = (const float*)d_in[11]; const float* be  = (const float*)d_in[12];
    const float* W3  = (const float*)d_in[13]; const float* b3  = (const float*)d_in[14];
    const float* W4  = (const float*)d_in[15]; const float* b4  = (const float*)d_in[16];

    float* out   = (float*)d_out;            // concat(recon, mu, Q, a, eigen)
    float* recon = out;
    float* muO   = out + (size_t)B * 784;
    float* qO    = out + (size_t)B * 884;
    float* aO    = out + (size_t)B * 2884;
    float* eO    = out + (size_t)B * 2889;

    char* w = (char*)d_ws;
    size_t off = 0;
    auto carve = [&](size_t bytes) {
        void* p = w + off;
        off = (off + bytes + 255) & ~(size_t)255;
        return p;
    };
    __bf16* xb   = (__bf16*)carve((size_t)B * K1P * 2);
    __bf16* h1b  = (__bf16*)carve((size_t)B * K2P * 2);
    __bf16* h3b  = (__bf16*)carve((size_t)B * K2P * 2);
    __bf16* zb   = (__bf16*)carve((size_t)B * 32 * 2);
    int*    idx  = (int*)carve((size_t)B * 4);
    __bf16* W1t  = (__bf16*)carve((size_t)400 * K1P * 2);
    __bf16* Wmut = (__bf16*)carve((size_t)100 * K2P * 2);
    __bf16* WQt  = (__bf16*)carve((size_t)2000 * K2P * 2);
    __bf16* Wet  = (__bf16*)carve((size_t)100 * K2P * 2);
    __bf16* W3t  = (__bf16*)carve((size_t)400 * 32 * 2);
    __bf16* W4t  = (__bf16*)carve((size_t)784 * K2P * 2);

    auto cb = [](int n) { return dim3((n + 255) / 256); };
    cvt_x_kernel<<<cb(B * K1P / 4), 256, 0, stream>>>(x, xb, B, 784, K1P);
    transpose_cvt_kernel<<<cb(400 * K1P), 256, 0, stream>>>(W1, W1t, 784, 400, K1P);
    transpose_cvt_kernel<<<cb(100 * K2P), 256, 0, stream>>>(Wmu, Wmut, 400, 100, K2P);
    transpose_cvt_kernel<<<cb(2000 * K2P), 256, 0, stream>>>(WQ, WQt, 400, 2000, K2P);
    transpose_cvt_kernel<<<cb(100 * K2P), 256, 0, stream>>>(We, Wet, 400, 100, K2P);
    transpose_cvt_kernel<<<cb(400 * 32), 256, 0, stream>>>(W3, W3t, 20, 400, 32);
    transpose_cvt_kernel<<<cb(784 * K2P), 256, 0, stream>>>(W4, W4t, 400, 784, K2P);

    dim3 blk(256);
    auto grid = [&](int N) { return dim3((N + BN - 1) / BN, B / BM); };

    // encoder
    gemm_bias_act<ACT_RELU_BF16><<<grid(400), blk, 0, stream>>>(xb, W1t, b1, h1b, B, 400, K1P, K2P);
    gemm_bias_act<ACT_NONE><<<grid(100), blk, 0, stream>>>(h1b, Wmut, bmu, muO, B, 100, K2P, 100);
    gemm_bias_act<ACT_EXP><<<grid(2000), blk, 0, stream>>>(h1b, WQt, bQ, qO, B, 2000, K2P, 2000);
    gemm_bias_act<ACT_EXP><<<grid(100), blk, 0, stream>>>(h1b, Wet, be, eO, B, 100, K2P, 100);

    // mixture pick + latent solve
    mixture_select_kernel<<<dim3((B * 32) / 256), blk, 0, stream>>>(h1b, Wa, ba, u, aO, idx, B, K2P);
    solve_z_kernel<<<dim3(B), dim3(32), 0, stream>>>(qO, muO, eO, idx, eps, zb);

    // decoder
    gemm_bias_act<ACT_RELU_BF16><<<grid(400), blk, 0, stream>>>(zb, W3t, b3, h3b, B, 400, 32, K2P);
    gemm_bias_act<ACT_SIGMOID><<<grid(784), blk, 0, stream>>>(h3b, W4t, b4, recon, B, 784, K2P, 784);
}
